// TransformerLayer_29291676959396
// MI455X (gfx1250) — matrix operand order repaired and checked by reading, not compiled, not run on hardware
//
#include <hip/hip_runtime.h>
#include <math.h>

typedef float  v2f  __attribute__((ext_vector_type(2)));
typedef float  v8f  __attribute__((ext_vector_type(8)));
typedef __bf16 v16bf __attribute__((ext_vector_type(16)));

#define D_MODEL 1024
#define D_FF    3072
#define SEQ     2048
#define NBATCH  2
#define NH      16
#define DH      64
#define M_TOT   (NBATCH*SEQ)   // 4096 tokens
#define QKV_N   (3*D_MODEL)    // 3072

// ---------------------------------------------------------------------------
// Matrix ops
// ---------------------------------------------------------------------------
__device__ __forceinline__ v8f wmma_f32(v2f a, v2f b, v8f c) {
    return __builtin_amdgcn_wmma_f32_16x16x4_f32(
        false, a, false, b, (short)0, c, false, false);
}
__device__ __forceinline__ v8f wmma_bf16(v16bf a, v16bf b, v8f c) {
    return __builtin_amdgcn_wmma_f32_16x16x32_bf16(
        false, a, false, b, (short)0, c, false, false);
}

// ---------------------------------------------------------------------------
// Split-bf16 packing: uint32 = (bf16_rne(x) << 16) | bf16_rne(x - hi)
// hi+lo represents x to ~2^-16 relative; 3 bf16 WMMAs recover ~fp32 GEMMs.
// ---------------------------------------------------------------------------
__device__ __forceinline__ unsigned pack_bf16x2(float x) {
    unsigned u = __float_as_uint(x);
    unsigned hi = (u + 0x7FFFu + ((u >> 16) & 1u)) & 0xFFFF0000u;  // rne bf16
    float r = x - __uint_as_float(hi);
    unsigned v = __float_as_uint(r);
    unsigned lo = (v + 0x7FFFu + ((v >> 16) & 1u)) >> 16;
    return hi | lo;
}

// 8 consecutive packed elems -> 4 dwords of bf16-hi pairs + 4 of bf16-lo pairs
// (v_perm_b32; pair dword = [elem2j+1 : elem2j] as required by WMMA operands)
__device__ __forceinline__ void unpack8(const unsigned* __restrict__ p,
                                        unsigned* H, unsigned* L) {
    const uint4* q = (const uint4*)p;
    uint4 a = q[0], b = q[1];
    H[0] = __builtin_amdgcn_perm(a.y, a.x, 0x07060302u);
    L[0] = __builtin_amdgcn_perm(a.y, a.x, 0x05040100u);
    H[1] = __builtin_amdgcn_perm(a.w, a.z, 0x07060302u);
    L[1] = __builtin_amdgcn_perm(a.w, a.z, 0x05040100u);
    H[2] = __builtin_amdgcn_perm(b.y, b.x, 0x07060302u);
    L[2] = __builtin_amdgcn_perm(b.y, b.x, 0x05040100u);
    H[3] = __builtin_amdgcn_perm(b.w, b.z, 0x07060302u);
    L[3] = __builtin_amdgcn_perm(b.w, b.z, 0x05040100u);
}

// ---------------------------------------------------------------------------
// Elementwise f32 -> packed split-bf16 (weights; once per launch)
// ---------------------------------------------------------------------------
__global__ __launch_bounds__(256)
void pack_kernel(const float* __restrict__ in, unsigned* __restrict__ out) {
    const int i = blockIdx.x * 1024 + threadIdx.x;
#pragma unroll
    for (int j = 0; j < 4; ++j) out[i + j * 256] = pack_bf16x2(in[i + j * 256]);
}

// ---------------------------------------------------------------------------
// RMS norm: 256-thread block per token row; output packed split-bf16
// (only GEMM-A consumers read rmsnorm output)
// ---------------------------------------------------------------------------
__global__ __launch_bounds__(256)
void rmsnorm_kernel(const float* __restrict__ x, const float* __restrict__ scale,
                    unsigned* __restrict__ outp) {
    __shared__ float red[8];
    const int row = blockIdx.x;
    const float* xr = x + (size_t)row * D_MODEL;
    float vals[4];
    float s = 0.f;
#pragma unroll
    for (int i = 0; i < 4; ++i) {
        vals[i] = xr[threadIdx.x + i * 256];
        s += vals[i] * vals[i];
    }
#pragma unroll
    for (int off = 16; off > 0; off >>= 1) s += __shfl_xor(s, off, 32);
    if ((threadIdx.x & 31) == 0) red[threadIdx.x >> 5] = s;
    __syncthreads();
    float tot = 0.f;
#pragma unroll
    for (int i = 0; i < 8; ++i) tot += red[i];
    const float inv = rsqrtf(tot * (1.0f / (float)D_MODEL) + 1e-6f);
    unsigned* orow = outp + (size_t)row * D_MODEL;
#pragma unroll
    for (int i = 0; i < 4; ++i) {
        const int c = threadIdx.x + i * 256;
        orow[c] = pack_bf16x2(vals[i] * scale[c] * inv);
    }
}

// ---------------------------------------------------------------------------
// Per-(token,head) cosine norm of q and k with sqrt(attn_scale[h]); in place.
// ---------------------------------------------------------------------------
__global__ __launch_bounds__(128)
void qknorm_kernel(float* __restrict__ qkv, const float* __restrict__ attn_scale) {
    const int lane = threadIdx.x & 31;
    const int w = threadIdx.x >> 5;
    const int g = blockIdx.x * 4 + w;
    const int m = g >> 5;
    const int rest = g & 31;
    const int which = rest >> 4;             // 0=q, 1=k
    const int h = rest & 15;
    float* p = qkv + (size_t)m * QKV_N + which * D_MODEL + h * DH + lane * 2;
    v2f v = *(v2f*)p;
    float s = v.x * v.x + v.y * v.y;
#pragma unroll
    for (int off = 16; off > 0; off >>= 1) s += __shfl_xor(s, off, 32);
    const float sc = sqrtf(attn_scale[h]) * rsqrtf(s + 1e-6f);
    v.x *= sc; v.y *= sc;
    *(v2f*)p = v;
}

// ---------------------------------------------------------------------------
// Split-bf16 GEMM: out[M,N] = A[M,K] @ W[N,K]^T (+ epilogue)
// A, W are packed split-bf16 (4B/elem, same traffic as f32).
// One wave = 64x64 tile; per 32-deep K slice: 32 b128 loads, 128 v_perm,
// 48 v_wmma_f32_16x16x32_bf16 (hh + hl + lh), fp32 accumulate.
// epi: 0 plain f32, 1 += aux (residual), 2 silu, 3 (c*aux) -> packed bf16x2
// ---------------------------------------------------------------------------
__global__ __launch_bounds__(32)
void gemm64_wmma(const unsigned* __restrict__ A, const unsigned* __restrict__ W,
                 float* __restrict__ out, const float* __restrict__ aux,
                 int N, int K, int epi) {
    const int lane = threadIdx.x & 31;
    const int l15 = lane & 15;
    const int hi  = lane >> 4;
    const int n0 = blockIdx.x * 64;
    const int m0 = blockIdx.y * 64;
    const unsigned* Ab = A + (size_t)(m0 + l15) * K;
    const unsigned* Wb = W + (size_t)(n0 + l15) * K;

    v8f zero = {};
    v8f c[4][4];
#pragma unroll
    for (int mt = 0; mt < 4; ++mt)
#pragma unroll
        for (int nt = 0; nt < 4; ++nt) c[mt][nt] = zero;

    for (int k = 0; k < K; k += 32) {
        // A fragments: K groups [k+8*hi, +8) and [k+16+8*hi, +8)
        v16bf ah[4], al[4];
#pragma unroll
        for (int mt = 0; mt < 4; ++mt) {
            union { unsigned d[8]; v16bf v; } H, L;
            const unsigned* pa = Ab + (size_t)mt * 16 * K + k;
            unpack8(pa + 8 * hi, H.d, L.d);
            unpack8(pa + 16 + 8 * hi, H.d + 4, L.d + 4);
            ah[mt] = H.v; al[mt] = L.v;
        }
        // B fragments: K groups [k+8*hi, +8) and [k+16+8*hi, +8)
#pragma unroll
        for (int nt = 0; nt < 4; ++nt) {
            union { unsigned d[8]; v16bf v; } H, L;
            const unsigned* pb = Wb + (size_t)nt * 16 * K + k + 8 * hi;
            unpack8(pb, H.d, L.d);
            unpack8(pb + 16, H.d + 4, L.d + 4);
            v16bf bh = H.v, bl = L.v;
#pragma unroll
            for (int mt = 0; mt < 4; ++mt) {
                c[mt][nt] = wmma_bf16(al[mt], bh, c[mt][nt]);
                c[mt][nt] = wmma_bf16(ah[mt], bl, c[mt][nt]);
                c[mt][nt] = wmma_bf16(ah[mt], bh, c[mt][nt]);
            }
        }
    }

#pragma unroll
    for (int mt = 0; mt < 4; ++mt)
#pragma unroll
        for (int nt = 0; nt < 4; ++nt)
#pragma unroll
            for (int v = 0; v < 8; ++v) {
                const int row = m0 + mt * 16 + v + 8 * hi;
                const int col = n0 + nt * 16 + l15;
                const size_t idx = (size_t)row * N + col;
                float val = c[mt][nt][v];
                if (epi == 1)      val += aux[idx];
                else if (epi == 2) val = val / (1.0f + __expf(-val));
                if (epi == 3) {
                    val *= aux[idx];
                    ((unsigned*)out)[idx] = pack_bf16x2(val);   // h, packed
                } else {
                    out[idx] = val;
                }
            }
}

// ---------------------------------------------------------------------------
// Flash attention (full-precision f32 WMMA), one wave per 16-row q tile.
// Output written packed split-bf16 (feeds the out-proj GEMM).
// ---------------------------------------------------------------------------
__global__ __launch_bounds__(128)
void attn_kernel(const float* __restrict__ qkv, unsigned* __restrict__ op) {
    __shared__ __align__(16) float Plds[4][16 * 18];
    const int lane = threadIdx.x & 31;
    const int w = threadIdx.x >> 5;
    const int l15 = lane & 15;
    const int hi  = lane >> 4;
    const int gw = blockIdx.x * 4 + w;
    const int qt = gw & 127;
    const int bh = gw >> 7;
    const int h = bh & 15;
    const int b = bh >> 4;

    const float* Qb = qkv + (size_t)(b * SEQ + qt * 16) * QKV_N + h * DH;
    const float* Kb = qkv + (size_t)(b * SEQ) * QKV_N + D_MODEL + h * DH;
    const float* Vb = qkv + (size_t)(b * SEQ) * QKV_N + 2 * D_MODEL + h * DH;

    v2f qa[16];
#pragma unroll
    for (int s = 0; s < 16; ++s)
        qa[s] = *(const v2f*)(Qb + (size_t)l15 * QKV_N + 4 * s + 2 * hi);

    v8f zero = {};
    v8f oacc[4];
#pragma unroll
    for (int nt = 0; nt < 4; ++nt) oacc[nt] = zero;
    float mrow[8], lrow[8];
#pragma unroll
    for (int v = 0; v < 8; ++v) { mrow[v] = -INFINITY; lrow[v] = 0.0f; }

    float* pl = &Plds[w][0];

    for (int kt = 0; kt < SEQ / 16; ++kt) {
        v8f S = zero;
#pragma unroll
        for (int s = 0; s < 16; ++s) {
            v2f kb = *(const v2f*)(Kb + (size_t)(kt * 16 + l15) * QKV_N + 4 * s + 2 * hi);
            S = wmma_f32(qa[s], kb, S);
        }
        float alpha[8];
#pragma unroll
        for (int v = 0; v < 8; ++v) {
            float mx = S[v];
            mx = fmaxf(mx, __shfl_xor(mx, 1, 32));
            mx = fmaxf(mx, __shfl_xor(mx, 2, 32));
            mx = fmaxf(mx, __shfl_xor(mx, 4, 32));
            mx = fmaxf(mx, __shfl_xor(mx, 8, 32));
            const float mnew = fmaxf(mrow[v], mx);
            const float p = __expf(S[v] - mnew);
            float rs = p;
            rs += __shfl_xor(rs, 1, 32);
            rs += __shfl_xor(rs, 2, 32);
            rs += __shfl_xor(rs, 4, 32);
            rs += __shfl_xor(rs, 8, 32);
            alpha[v] = __expf(mrow[v] - mnew);
            lrow[v] = lrow[v] * alpha[v] + rs;
            mrow[v] = mnew;
            S[v] = p;
        }
#pragma unroll
        for (int nt = 0; nt < 4; ++nt)
#pragma unroll
            for (int v = 0; v < 8; ++v) oacc[nt][v] *= alpha[v];

#pragma unroll
        for (int v = 0; v < 8; ++v)
            pl[(v + 8 * hi) * 18 + l15] = S[v];
        __syncthreads();

#pragma unroll
        for (int s2 = 0; s2 < 4; ++s2) {
            v2f pa = *(const v2f*)(pl + l15 * 18 + 4 * s2 + 2 * hi);
            const int key = kt * 16 + 4 * s2 + 2 * hi;
#pragma unroll
            for (int nt = 0; nt < 4; ++nt) {
                v2f vb;
                vb.x = Vb[(size_t)key * QKV_N + nt * 16 + l15];
                vb.y = Vb[(size_t)(key + 1) * QKV_N + nt * 16 + l15];
                oacc[nt] = wmma_f32(pa, vb, oacc[nt]);
            }
        }
        __syncthreads();
    }

#pragma unroll
    for (int nt = 0; nt < 4; ++nt)
#pragma unroll
        for (int v = 0; v < 8; ++v) {
            const int row = b * SEQ + qt * 16 + v + 8 * hi;
            const int col = h * DH + nt * 16 + l15;
            op[(size_t)row * D_MODEL + col] = pack_bf16x2(oacc[nt][v] / lrow[v]);
        }
}

// ---------------------------------------------------------------------------
extern "C" void kernel_launch(void* const* d_in, const int* in_sizes, int n_in,
                              void* d_out, int out_size, void* d_ws, size_t ws_size,
                              hipStream_t stream) {
    const float* x             = (const float*)d_in[0];
    // d_in[1] = pos (unused by the reference)
    const float* norm_scale    = (const float*)d_in[2];
    const float* w_qkv         = (const float*)d_in[3];
    const float* attn_scale    = (const float*)d_in[4];
    const float* w_out         = (const float*)d_in[5];
    const float* ff_norm_scale = (const float*)d_in[6];
    const float* w_up          = (const float*)d_in[7];
    const float* w_down        = (const float*)d_in[8];
    float* out = (float*)d_out;

    // Workspace layout (4-byte elements); ~138.4 MiB total.
    unsigned* base = (unsigned*)d_ws;
    unsigned* actp    = base;                 // 4.2M  : xn_p -> o_p -> xn2_p (reused)
    float*    qkv     = (float*)(base + 4194304);   // 12.6M : qkv f32 -> sg f32 -> h packed
    float*    x1      = (float*)(base + 16777216);  // 4.2M  : post-attention residual
    unsigned* wqkv_p  = base + 20971520;      // 3.15M
    unsigned* wout_p  = base + 24117248;      // 1.05M
    unsigned* wup_p   = base + 25165824;      // 6.29M
    unsigned* wdown_p = base + 31457280;      // 3.15M
    float*    sg      = qkv;
    unsigned* hpk     = (unsigned*)qkv;

    // 0) pack weights to split-bf16 (recomputed each call; deterministic)
    pack_kernel<<<(QKV_N * D_MODEL) / 1024, 256, 0, stream>>>(w_qkv, wqkv_p);
    pack_kernel<<<(D_MODEL * D_MODEL) / 1024, 256, 0, stream>>>(w_out, wout_p);
    pack_kernel<<<(2 * D_FF * D_MODEL) / 1024, 256, 0, stream>>>(w_up, wup_p);
    pack_kernel<<<(D_MODEL * D_FF) / 1024, 256, 0, stream>>>(w_down, wdown_p);

    // 1) xn = rms_norm(x) -> packed
    rmsnorm_kernel<<<M_TOT, 256, 0, stream>>>(x, norm_scale, actp);
    // 2) qkv = xn @ w_qkv^T (f32 out)
    gemm64_wmma<<<dim3(QKV_N / 64, M_TOT / 64), 32, 0, stream>>>(
        actp, wqkv_p, qkv, nullptr, QKV_N, D_MODEL, 0);
    // 3) cosine-normalize q,k per head (carries sqrt(attn_scale))
    qknorm_kernel<<<(M_TOT * NH * 2) / 4, 128, 0, stream>>>(qkv, attn_scale);
    // 4) o = flash_attention(q,k,v) -> packed into actp
    attn_kernel<<<(NBATCH * NH * (SEQ / 16)) / 4, 128, 0, stream>>>(qkv, actp);
    // 5) x1 = o @ w_out^T + x
    gemm64_wmma<<<dim3(D_MODEL / 64, M_TOT / 64), 32, 0, stream>>>(
        actp, wout_p, x1, x, D_MODEL, D_MODEL, 1);
    // 6) xn2 = rms_norm(x1) -> packed into actp
    rmsnorm_kernel<<<M_TOT, 256, 0, stream>>>(x1, ff_norm_scale, actp);
    // 7) sg = silu(xn2 @ w_up[dff:,:]^T) (f32)
    gemm64_wmma<<<dim3(D_FF / 64, M_TOT / 64), 32, 0, stream>>>(
        actp, wup_p + (size_t)D_FF * D_MODEL, sg, nullptr, D_FF, D_MODEL, 2);
    // 8) h = (xn2 @ w_up[:dff,:]^T) * sg -> packed, in place over sg
    gemm64_wmma<<<dim3(D_FF / 64, M_TOT / 64), 32, 0, stream>>>(
        actp, wup_p, sg, sg, D_FF, D_MODEL, 3);
    // 9) out = h @ w_down^T + x1
    gemm64_wmma<<<dim3(D_MODEL / 64, M_TOT / 64), 32, 0, stream>>>(
        hpk, wdown_p, out, x1, D_MODEL, D_FF, 1);
}
